// SNN_36550171689497
// MI455X (gfx1250) — matrix operand order repaired and checked by reading, not compiled, not run on hardware
//
#include <hip/hip_runtime.h>
#include <cstdint>

// ---------------------------------------------------------------------------
// Spiking CNN forward (conv-BN-LIF-pool x2 + FC-LIF readout) for gfx1250.
//  * all GEMMs (implicit-GEMM convs + FC) -> v_wmma_f32_16x16x32_f16 (wave32)
//  * BN training stats fused into the conv kernels (shfl + LDS + f32 atomics)
//    -- removes two full re-read passes (~385 MB of HBM traffic)
//  * NHWC intermediates: each lane's 8 C-fragment values are contiguous ->
//    global_store_b128 pairs; BN/LIF/pool kernel is channel-fastest coalesced
//  * weights staged to LDS via GLOBAL_LOAD_ASYNC_TO_LDS_B32 (+ s_wait_asynccnt)
// ---------------------------------------------------------------------------

typedef __attribute__((ext_vector_type(16))) _Float16 v16h;
typedef __attribute__((ext_vector_type(8)))  float    v8f;

#define T_STEPS 20
#define BATCH   256
#define TN      (T_STEPS * BATCH)          // 5120 merged (t,n) images
#define HW1     784                         // 28*28
#define HW2     196                         // 14*14
#define HW3     49                          // 7*7

// --- async global->LDS staging (CDNA5), guarded ----------------------------
#if defined(__HIP_DEVICE_COMPILE__)
#if __has_builtin(__builtin_amdgcn_global_load_async_to_lds_b32)
#define USE_ASYNC_G2L 1
#endif
#endif

#ifdef USE_ASYNC_G2L
typedef __attribute__((address_space(1))) int GI;   // global (prints __device__)
typedef __attribute__((address_space(3))) int LI;   // LDS    (prints __shared__)
__device__ __forceinline__ void g2l_b32(const float* g, float* l) {
    // generic LDS pointers carry the LDS byte offset in their low 32 bits
    __builtin_amdgcn_global_load_async_to_lds_b32(
        (GI*)(uintptr_t)g,
        (LI*)(uintptr_t)(uint32_t)(uintptr_t)l, 0, 0);
}
__device__ __forceinline__ void g2l_wait() {
#if __has_builtin(__builtin_amdgcn_s_wait_asynccnt)
    __builtin_amdgcn_s_wait_asynccnt(0);
#else
    asm volatile("s_wait_asynccnt 0x0" ::: "memory");
#endif
}
#endif

// stage n floats of weights into LDS (all threads), barrier on exit
__device__ __forceinline__ void stage_weights(const float* g, float* lds,
                                              int n, int tid, int nthr) {
#ifdef USE_ASYNC_G2L
    for (int i = tid; i < n; i += nthr) g2l_b32(g + i, lds + i);
    g2l_wait();
#else
    for (int i = tid; i < n; i += nthr) lds[i] = g[i];
#endif
    __syncthreads();
}

// --- WMMA 16x16x32 f16 (lane, i) -> K used here: contiguous k = 16*(lane>>4) + i, same for A and B ------
__device__ __forceinline__ int a_k_of(int lane, int i) {
    return i + ((lane >> 4) << 4);
}
__device__ __forceinline__ int b_k_of(int lane, int i) {
    return i + ((lane >> 4) << 4);
}
// C/D f32 16x16: element j of lane L -> M = j + 8*(L>=16), N = L&15

// per-wave BN stats from the C fragment: shfl-reduce each channel over the
// 16 positions, LDS-atomic accumulate (block arrays are per-block channels)
__device__ __forceinline__ void wave_stats_accum(const v8f& acc, int lane,
                                                 float* ls_sum, float* ls_sq) {
    #pragma unroll
    for (int j = 0; j < 8; ++j) {
        float s = acc[j];
        float q = acc[j] * acc[j];
        #pragma unroll
        for (int m = 1; m < 16; m <<= 1) {
            s += __shfl_xor(s, m, 32);
            q += __shfl_xor(q, m, 32);
        }
        if ((lane & 15) == 0) {
            int ch = j + ((lane >> 4) << 3);
            atomicAdd(&ls_sum[ch], s);   // ds_add_f32
            atomicAdd(&ls_sq[ch],  q);
        }
    }
}

// ---------------------------------------------------------------------------
// Stage 1: conv1 3->16, 3x3 SAME, implicit GEMM. K = 27 -> 32, one WMMA/tile.
// Output NHWC f32 [TN*784][16]; fused per-channel sum/sumsq for BN1.
// ---------------------------------------------------------------------------
__global__ __launch_bounds__(256)
void conv1_wmma(const float* __restrict__ x,   // [N,T,3,28,28]
                const float* __restrict__ w1,  // [16,3,3,3]
                float* __restrict__ out,       // [TN*784][16]
                float* __restrict__ stats)     // [16 sum][16 sumsq]
{
    __shared__ float lw[16 * 27];
    __shared__ float ls_sum[16], ls_sq[16];
    const int tid = threadIdx.x;
    if (tid < 16) { ls_sum[tid] = 0.f; ls_sq[tid] = 0.f; }
    stage_weights(w1, lw, 16 * 27, tid, 256);          // async -> LDS + barrier

    const int lane = tid & 31;
    const int wave = tid >> 5;
    const int tile = blockIdx.x * 8 + wave;
    const int p0   = tile * 16;

    // A fragment: padded weights from LDS
    const int m = lane & 15;
    v16h a;
    #pragma unroll
    for (int i = 0; i < 16; ++i) {
        int k = a_k_of(lane, i);
        a[i] = (k < 27) ? (_Float16)lw[m * 27 + k] : (_Float16)0.0f;
    }

    // B fragment: im2col patch for position p0 + (lane&15)
    const int n   = lane & 15;
    const int p   = p0 + n;
    const int img = p / HW1;
    const int hw  = p % HW1;
    const int h   = hw / 28, w = hw % 28;
    const int t   = img / BATCH, nb = img % BATCH;
    const float* xb = x + ((size_t)nb * T_STEPS + t) * (3 * HW1);
    v16h b;
    #pragma unroll
    for (int i = 0; i < 16; ++i) {
        int k = b_k_of(lane, i);
        float v = 0.0f;
        if (k < 27) {
            int ci = k / 9, r = k % 9;
            int hh = h + r / 3 - 1, ww = w + r % 3 - 1;
            if (hh >= 0 && hh < 28 && ww >= 0 && ww < 28)
                v = xb[ci * HW1 + hh * 28 + ww];
        }
        b[i] = (_Float16)v;
    }

    v8f acc = {};
    acc = __builtin_amdgcn_wmma_f32_16x16x32_f16(false, a, false, b,
                                                 (short)0, acc, false, false);

    // NHWC store: 8 contiguous channel values -> two b128 stores
    float4* ob = (float4*)(out + ((size_t)p * 16 + ((lane >> 4) << 3)));
    ob[0] = make_float4(acc[0], acc[1], acc[2], acc[3]);
    ob[1] = make_float4(acc[4], acc[5], acc[6], acc[7]);

    wave_stats_accum(acc, lane, ls_sum, ls_sq);
    __syncthreads();
    if (tid < 16) {
        atomicAdd(&stats[tid],      ls_sum[tid]);      // global_atomic_add_f32
        atomicAdd(&stats[16 + tid], ls_sq[tid]);
    }
}

// ---------------------------------------------------------------------------
// Fused BN (training stats) + LIF (tau=2, vth=1, hard reset) + 2x2 maxpool.
// Channel-fastest thread mapping for fully coalesced NHWC access.
// in: f32 [TN*Hin*Hin][C], out: f16 [TN*Hp*Hp][C]
// ---------------------------------------------------------------------------
__global__ __launch_bounds__(256)
void bn_lif_pool(const float* __restrict__ in, const float* __restrict__ stats,
                 const float* __restrict__ gamma, const float* __restrict__ beta,
                 _Float16* __restrict__ outp, int C, int Hin, int Hp, float invcnt)
{
    int idx = blockIdx.x * blockDim.x + threadIdx.x;
    const int Win = Hin, Wp = Hp;
    int c  = idx % C;  int r = idx / C;
    int wp = r % Wp;   r /= Wp;
    int hp = r % Hp;   r /= Hp;
    int n  = r;

    float mean = stats[c] * invcnt;
    float var  = stats[C + c] * invcnt - mean * mean;
    float sc   = gamma[c] * rsqrtf(var + 1e-5f);
    float sh   = beta[c] - mean * sc;

    const int h0 = hp * 2, w0 = wp * 2;
    const int pa = h0 * Win + w0, pb = pa + 1, pcx = pa + Win, pd = pcx + 1;

    float v0 = 0.f, v1 = 0.f, v2 = 0.f, v3 = 0.f;
    for (int t = 0; t < T_STEPS; ++t) {
        size_t ib = (size_t)(t * BATCH + n) * (Hin * Win);
        if (t + 1 < T_STEPS) {  // prefetch next time-step plane (strided walk)
            size_t ibn = (size_t)((t + 1) * BATCH + n) * (Hin * Win);
            __builtin_prefetch(&in[(ibn + pa) * C + c], 0, 1);
        }
        float x0 = in[(ib + pa)  * C + c] * sc + sh;
        float x1 = in[(ib + pb)  * C + c] * sc + sh;
        float x2 = in[(ib + pcx) * C + c] * sc + sh;
        float x3 = in[(ib + pd)  * C + c] * sc + sh;
        v0 = 0.5f * (v0 + x0);  v1 = 0.5f * (v1 + x1);   // v += (x-v)/tau
        v2 = 0.5f * (v2 + x2);  v3 = 0.5f * (v3 + x3);
        float s0 = (v0 >= 1.f) ? 1.f : 0.f;
        float s1 = (v1 >= 1.f) ? 1.f : 0.f;
        float s2 = (v2 >= 1.f) ? 1.f : 0.f;
        float s3 = (v3 >= 1.f) ? 1.f : 0.f;
        v0 = (s0 != 0.f) ? 0.f : v0;  v1 = (s1 != 0.f) ? 0.f : v1;  // hard reset
        v2 = (s2 != 0.f) ? 0.f : v2;  v3 = (s3 != 0.f) ? 0.f : v3;
        float sp = fmaxf(fmaxf(s0, s1), fmaxf(s2, s3));             // maxpool
        outp[((size_t)(t * BATCH + n) * (Hp * Wp) + hp * Wp + wp) * C + c] =
            (_Float16)sp;
    }
}

// ---------------------------------------------------------------------------
// Stage 3: conv2 16->32, 3x3 SAME on 14x14 spike maps. K = 144 -> 160 =
// 5 chained WMMAs/tile. grid.y = output-channel half. Fused BN2 stats.
// ---------------------------------------------------------------------------
__global__ __launch_bounds__(256)
void conv2_wmma(const _Float16* __restrict__ sp,  // [TN*196][16] spikes
                const float* __restrict__ w2,     // [32,16,3,3]
                float* __restrict__ out,          // [TN*196][32]
                float* __restrict__ stats)        // [32 sum][32 sumsq]
{
    __shared__ float lw[16 * 144];
    __shared__ float ls_sum[16], ls_sq[16];
    const int tid = threadIdx.x;
    const int cb  = blockIdx.y;                    // channel block 0/1
    if (tid < 16) { ls_sum[tid] = 0.f; ls_sq[tid] = 0.f; }
    stage_weights(w2 + (size_t)cb * 16 * 144, lw, 16 * 144, tid, 256);

    const int lane = tid & 31;
    const int wave = tid >> 5;
    const int tile = blockIdx.x * 8 + wave;
    const int p0   = tile * 16;
    const int m    = lane & 15;

    const int n   = lane & 15;
    const int p   = p0 + n;
    const int img = p / HW2;
    const int hw  = p % HW2;
    const int h   = hw / 14, w = hw % 14;

    v8f acc = {};
    #pragma unroll
    for (int kk = 0; kk < 5; ++kk) {
        const int kb = kk * 32;
        v16h a, b;
        #pragma unroll
        for (int i = 0; i < 16; ++i) {
            int k = kb + a_k_of(lane, i);
            a[i] = (k < 144) ? (_Float16)lw[m * 144 + k] : (_Float16)0.0f;
        }
        #pragma unroll
        for (int i = 0; i < 16; ++i) {
            int k = kb + b_k_of(lane, i);
            _Float16 v = (_Float16)0.0f;
            if (k < 144) {
                int ci = k / 9, r = k % 9;
                int hh = h + r / 3 - 1, ww = w + r % 3 - 1;
                if (hh >= 0 && hh < 14 && ww >= 0 && ww < 14)
                    v = sp[((size_t)img * HW2 + hh * 14 + ww) * 16 + ci];
            }
            b[i] = v;
        }
        acc = __builtin_amdgcn_wmma_f32_16x16x32_f16(false, a, false, b,
                                                     (short)0, acc, false, false);
    }

    float4* ob = (float4*)(out + ((size_t)p * 32 + cb * 16 + ((lane >> 4) << 3)));
    ob[0] = make_float4(acc[0], acc[1], acc[2], acc[3]);
    ob[1] = make_float4(acc[4], acc[5], acc[6], acc[7]);

    wave_stats_accum(acc, lane, ls_sum, ls_sq);
    __syncthreads();
    if (tid < 16) {
        atomicAdd(&stats[cb * 16 + tid],      ls_sum[tid]);
        atomicAdd(&stats[32 + cb * 16 + tid], ls_sq[tid]);
    }
}

// ---------------------------------------------------------------------------
// Stage 5: FC 1568 -> 8 (padded to 16 rows). K = 1568 = 49 * 32 exactly.
// feat NHWC [img*49 + hw3][32]; feature index f = c*49 + hw3 (NCHW flatten).
// ---------------------------------------------------------------------------
__global__ __launch_bounds__(256)
void fc_wmma(const _Float16* __restrict__ feat, const float* __restrict__ wfc,
             float* __restrict__ logits)        // [TN][8]
{
    __shared__ float lw[8 * 1568];               // ~50 KB (WGP has 320 KB LDS)
    const int tid = threadIdx.x;
    stage_weights(wfc, lw, 8 * 1568, tid, 256);

    const int lane = tid & 31;
    const int wave = tid >> 5;
    const int tile = blockIdx.x * 8 + wave;
    const int p0   = tile * 16;
    const int m    = lane & 15;
    const int img  = p0 + (lane & 15);

    v8f acc = {};
    for (int kk = 0; kk < 49; ++kk) {
        const int kb = kk * 32;
        v16h a, b;
        #pragma unroll
        for (int i = 0; i < 16; ++i) {
            int k = kb + a_k_of(lane, i);
            a[i] = (m < 8) ? (_Float16)lw[m * 1568 + k] : (_Float16)0.0f;
        }
        #pragma unroll
        for (int i = 0; i < 16; ++i) {
            int k = kb + b_k_of(lane, i);
            int c = k / HW3, hw3 = k % HW3;
            b[i] = feat[((size_t)img * HW3 + hw3) * 32 + c];
        }
        acc = __builtin_amdgcn_wmma_f32_16x16x32_f16(false, a, false, b,
                                                     (short)0, acc, false, false);
    }
    if (lane < 16) {                 // lanes 0-15 hold M=0..7 (valid classes)
        float4* ob = (float4*)(logits + (size_t)img * 8);
        ob[0] = make_float4(acc[0], acc[1], acc[2], acc[3]);
        ob[1] = make_float4(acc[4], acc[5], acc[6], acc[7]);
    }
}

// ---------------------------------------------------------------------------
// Final LIF over logits + firing-rate mean -> out [N, 8]
// ---------------------------------------------------------------------------
__global__ __launch_bounds__(256)
void lif_readout(const float* __restrict__ logits, float* __restrict__ out)
{
    int idx = blockIdx.x * blockDim.x + threadIdx.x;   // 2048 = 256*8
    int n = idx >> 3, o = idx & 7;
    float v = 0.f, accum = 0.f;
    for (int t = 0; t < T_STEPS; ++t) {
        float x = logits[(size_t)(t * BATCH + n) * 8 + o];
        v = 0.5f * (v + x);
        float s = (v >= 1.f) ? 1.f : 0.f;
        accum += s;
        v = (s != 0.f) ? 0.f : v;
    }
    out[n * 8 + o] = accum * (1.f / T_STEPS);
}

// ---------------------------------------------------------------------------
extern "C" void kernel_launch(void* const* d_in, const int* in_sizes, int n_in,
                              void* d_out, int out_size, void* d_ws, size_t ws_size,
                              hipStream_t stream)
{
    (void)in_sizes; (void)n_in; (void)out_size; (void)ws_size;
    const float* x   = (const float*)d_in[0];
    const float* w1  = (const float*)d_in[1];
    const float* g1  = (const float*)d_in[2];
    const float* b1  = (const float*)d_in[3];
    const float* w2  = (const float*)d_in[4];
    const float* g2  = (const float*)d_in[5];
    const float* b2  = (const float*)d_in[6];
    const float* wfc = (const float*)d_in[7];
    float* out = (float*)d_out;

    // Workspace layout (aliased, ~290 MB):
    //   [0,128)    stats1: sum[16], sumsq[16]
    //   [128,384)  stats2: sum[32], sumsq[32]
    //   regionA @512 : conv1 out f32 NHWC (257MB), reused as conv2 out (128MB)
    //   regionB      : spikes1 f16 (32MB), reused as spikes2 f16 (16MB)
    //   buf5         : logits f32 (160KB) after regionB
    char* ws = (char*)d_ws;
    float*    stats1 = (float*)ws;
    float*    stats2 = (float*)(ws + 128);
    float*    buf1   = (float*)(ws + 512);
    size_t    sizeA  = (size_t)TN * 16 * HW1 * sizeof(float);
    char*     regB   = ws + 512 + sizeA;
    _Float16* buf2   = (_Float16*)regB;          // spikes after layer 1
    float*    buf3   = buf1;                     // conv2 out reuses regionA
    _Float16* buf4   = (_Float16*)regB;          // spikes after layer 2
    float*    buf5   = (float*)(regB + (size_t)TN * 16 * HW2 * sizeof(_Float16));

    (void)hipMemsetAsync(d_ws, 0, 512, stream);  // zero BN stat accumulators

    // conv1: TN*784/16 = 250880 tiles, 8 waves/block (stats fused)
    conv1_wmma<<<31360, 256, 0, stream>>>(x, w1, buf1, stats1);
    // BN1 + LIF + pool: 256*16*14*14 / 256 = 3136 blocks
    bn_lif_pool<<<3136, 256, 0, stream>>>(buf1, stats1, g1, b1, buf2,
                                          16, 28, 14, 1.f / ((float)TN * HW1));
    // conv2: TN*196/16 = 62720 tiles -> 7840 blocks x 2 channel halves
    conv2_wmma<<<dim3(7840, 2), 256, 0, stream>>>(buf2, w2, buf3, stats2);
    // BN2 + LIF + pool: 256*32*7*7 / 256 = 1568 blocks
    bn_lif_pool<<<1568, 256, 0, stream>>>(buf3, stats2, g2, b2, buf4,
                                          32, 14, 7, 1.f / ((float)TN * HW2));
    // FC: 5120/16 = 320 tiles -> 40 blocks
    fc_wmma<<<40, 256, 0, stream>>>(buf4, wfc, buf5);
    // LIF + rate readout: 2048 threads
    lif_readout<<<8, 256, 0, stream>>>(buf5, out);
}
